// FMLayer_1099511628175
// MI455X (gfx1250) — matrix operand order repaired and checked by reading, not compiled, not run on hardware
//
#include <hip/hip_runtime.h>
#include <stdint.h>

typedef __attribute__((ext_vector_type(16))) __bf16 v16bf;
typedef __attribute__((ext_vector_type(8)))  float  v8f;

#define N_DIM   4096
#define K_DIM   256
#define BATCH   8192
#define KCHUNKS (N_DIM / 32)   // 128 k-chunks of 32
#define NTILES  (K_DIM / 16)   // 16 n-tiles of 16

// ---------------------------------------------------------------------------
// Prep: swizzle V (4096x256 f32 row-major) into bf16 WMMA B-fragments.
// Layout: vb[nt][kc][lane][half], lane = N (lane&15) + 16 * k-half (0..31), half = k slot in table order (0..15).
// Each lane's fragment is 32 contiguous bytes -> single coalesced load later.
// ---------------------------------------------------------------------------
__global__ void fm_prep_vb(const float* __restrict__ V, __bf16* __restrict__ vb) {
    int tid  = blockIdx.x * blockDim.x + threadIdx.x;     // 65536 threads total
    int lane = tid & 31;
    int kc   = (tid >> 5) & (KCHUNKS - 1);
    int nt   = tid >> 12;                                 // tid >> (5+7)
    int k    = kc * 32 + ((lane >> 4) << 3);
    int n0   = nt * 16 + (lane & 15);
    const float* src = V + (size_t)k * K_DIM + n0;        // 16 k-strided f32 (table k order)
    __bf16* dst = vb + ((size_t)(nt * KCHUNKS + kc) * 32 + lane) * 16;
#pragma unroll
    for (int i = 0; i < 16; ++i) dst[i] = (__bf16)src[(size_t)(i + ((i >> 3) << 3)) * K_DIM];
}

// ---------------------------------------------------------------------------
// Main: per workgroup = 16 rows of x, all 256 output columns.
// out[r] = x[r]·w + b + sum_k (x[r]·V[:,k])^2
// ---------------------------------------------------------------------------
__global__ void __launch_bounds__(256) fm_main(const float*  __restrict__ x,
                                               const __bf16* __restrict__ vb,
                                               const float*  __restrict__ w,
                                               const float*  __restrict__ bptr,
                                               float*        __restrict__ out) {
    __shared__ __attribute__((aligned(32))) __bf16 a_stage[32][16];
    __shared__ float row_acc[16];

    const int t    = threadIdx.x;
    const int lane = t & 31;
    const int wave = t >> 5;                // 0..7
    const int row0 = blockIdx.x * 16;

    if (t < 16) row_acc[t] = 0.0f;

    // --- staging coordinates: thread t stages x[row0+sm][kc*32 + 2*sp .. +1] ---
    const int sm = t >> 4;                  // row in tile, 0..15
    const int sp = t & 15;                  // pair index, 0..15
    const int kk = 2 * sp;                  // kk, kk+1 within chunk
    // A-fragment layout for 16x16x32 bf16 (ISA 7.12.2):
    //   lane = m + 16*((kk>>3)&1), half = (kk&7) + 8*(kk>>4)
    const int s_lane = sm + (((kk >> 3) & 1) << 4);
    const int s_half = (kk & 7) + ((kk >> 4) << 3);

    const float* xrow = x + (size_t)(row0 + sm) * N_DIM + kk;
    const float* wcol = w + kk;

    const v16bf* bfr = (const v16bf*)vb;
    const int nt0 = wave * 2;
    const int nt1 = nt0 + 1;

    v8f acc0 = {};
    v8f acc1 = {};
    float wpart = 0.0f;

    for (int kc = 0; kc < KCHUNKS; ++kc) {
        // ---- stage A tile: f32 -> bf16, fragment order, + fold in w·x ----
        float x0 = xrow[0];
        float x1 = xrow[1];
        __builtin_prefetch(xrow + 32, 0, 0);        // next chunk of this row
        wpart = fmaf(x0, wcol[0], wpart);
        wpart = fmaf(x1, wcol[1], wpart);
        union { struct { __bf16 lo, hi; } h; uint32_t u; } pk;
        pk.h.lo = (__bf16)x0;
        pk.h.hi = (__bf16)x1;
        *(uint32_t*)&a_stage[s_lane][s_half] = pk.u;   // s_half is even -> 4B aligned
        xrow += 32;
        wcol += 32;
        __syncthreads();

        // ---- WMMA: A shared by all 8 waves, B from swizzled V in d_ws ----
        v16bf a  = *(const v16bf*)&a_stage[lane][0];
        v16bf b0 = bfr[(size_t)(nt0 * KCHUNKS + kc) * 32 + lane];
        v16bf b1 = bfr[(size_t)(nt1 * KCHUNKS + kc) * 32 + lane];
        acc0 = __builtin_amdgcn_wmma_f32_16x16x32_bf16(false, a, false, b0,
                                                       (short)0, acc0, false, false);
        acc1 = __builtin_amdgcn_wmma_f32_16x16x32_bf16(false, a, false, b1,
                                                       (short)0, acc1, false, false);
        __syncthreads();   // WAR: wmma reads done before next stage overwrites LDS
    }

    // ---- interaction: row-wise sum of z^2 ----
    // C layout: VGPR v holds (lanes 0-15 -> M=v, N=lane) (lanes 16-31 -> M=v+8, N=lane-16)
#pragma unroll
    for (int v = 0; v < 8; ++v) {
        float s = acc0[v] * acc0[v] + acc1[v] * acc1[v];
        s += __shfl_xor(s, 1);
        s += __shfl_xor(s, 2);
        s += __shfl_xor(s, 4);
        s += __shfl_xor(s, 8);
        if ((lane & 15) == 0)
            atomicAdd(&row_acc[v + ((lane >> 4) << 3)], s);   // ds_add_f32
    }
    atomicAdd(&row_acc[sm], wpart);                            // linear term
    __syncthreads();

    if (t < 16) out[row0 + t] = row_acc[t] + bptr[0];
}

// ---------------------------------------------------------------------------
extern "C" void kernel_launch(void* const* d_in, const int* in_sizes, int n_in,
                              void* d_out, int out_size, void* d_ws, size_t ws_size,
                              hipStream_t stream) {
    (void)in_sizes; (void)n_in; (void)out_size; (void)ws_size;
    const float* x = (const float*)d_in[0];   // (8192, 4096)
    const float* V = (const float*)d_in[1];   // (4096, 256)
    const float* w = (const float*)d_in[2];   // (4096,)
    const float* b = (const float*)d_in[3];   // (1,)
    __bf16* vb = (__bf16*)d_ws;               // 16*128*32*16*2 = 2 MB scratch

    fm_prep_vb<<<(NTILES * KCHUNKS * 32) / 256, 256, 0, stream>>>(V, vb);
    fm_main<<<BATCH / 16, 256, 0, stream>>>(x, vb, w, b, (float*)d_out);
}
